// MoE_82987358094102
// MI455X (gfx1250) — matrix operand order repaired and checked by reading, not compiled, not run on hardware
//
#include <hip/hip_runtime.h>

// ---------------- problem constants ----------------
#define E_    8
#define TOPK_ 2
#define NEMB  1024
#define NHID  2048
#define NB_   8
#define NT_   2048
#define CAP_  640
#define ME_   (NB_ * CAP_)     // 5120 rows per expert
#define NTOK  (NB_ * NT_)      // 16384 tokens

typedef unsigned short u16;
typedef __attribute__((ext_vector_type(16))) __bf16 bf16x16;
typedef __attribute__((ext_vector_type(8)))  float  f32x8;
typedef __attribute__((ext_vector_type(4))) unsigned int u32x4;
typedef __attribute__((ext_vector_type(4))) int i32x4;
typedef __attribute__((ext_vector_type(8))) int i32x8;

#if defined(__has_include)
#if __has_include(<hip/amd_detail/amd_gfx1250_TDM.h>)
#define TDM_ARITY6 1
#endif
#endif

union FragU { bf16x16 bf; unsigned int u[8]; };

__device__ __forceinline__ u16 f2bf(float f) {
  unsigned int u = __float_as_uint(f);
  u += 0x7fffu + ((u >> 16) & 1u);   // round-to-nearest-even
  return (u16)(u >> 16);
}

__device__ __forceinline__ f32x8 wmma_bf16(bf16x16 a, bf16x16 b, f32x8 c) {
  return __builtin_amdgcn_wmma_f32_16x16x32_bf16(false, a, false, b, (short)0, c, false, false);
}

// A fragment (16x32 bf16) from LDS, row-major, strideU ushorts per row.
// ISA layout: m = lane%16 ; element i -> k = (i/8)*16 + (lane/16)*8 + (i%8)
__device__ __forceinline__ bf16x16 load_a_frag(const u16* s, int m0, int k0, int strideU, int lane) {
  const int m = m0 + (lane & 15);
  const int khalf = (lane >> 4) << 3;
  const unsigned int* p = (const unsigned int*)(s + (size_t)m * strideU);
  FragU fr;
#pragma unroll
  for (int j = 0; j < 8; ++j) {
    const int klo = k0 + ((j >> 2) << 4) + khalf + ((j & 3) << 1);
    fr.u[j] = p[klo >> 1];
  }
  return fr.bf;
}

// B fragment (32x16 bf16) from LDS stage holding B^T tile [n][k], strideU ushorts/row.
// ISA layout: n = lane%16 ; element i -> k = (i/8)*16 + (lane/16)*8 + (i%8)
__device__ __forceinline__ bf16x16 load_b_frag(const u16* s, int n0, int strideU, int lane) {
  const int n = n0 + (lane & 15);
  const unsigned int* p = (const unsigned int*)(s + (size_t)n * strideU);
  const int kq = (lane >> 4) << 2;  // dword offset 0 or 4
  FragU fr;
#pragma unroll
  for (int j = 0; j < 8; ++j) fr.u[j] = p[kq + ((j >> 2) << 3) + (j & 3)];
  return fr.bf;
}

// ---------------- Tensor Data Mover: 2-D tile load, rows of 32 bf16 elems ----------------
// D# per CDNA5 ISA: tile_dim0=32 elems (16 dwords),
// pad_enable with pad_interval=3 (16 dwords) + pad_amount=3 (4 dwords)
// -> LDS rows land with stride 40 ushorts (bank-conflict-free).
__device__ __forceinline__ void tdm_load_2d(const void* gsrc, unsigned lds_byte_off,
                                            int tile_rows, unsigned stride_elems,
                                            unsigned tensor_rows) {
  const unsigned long long ga = (unsigned long long)(uintptr_t)gsrc;
  u32x4 g0;
  g0[0] = 1u;                                            // count=1, user D#
  g0[1] = lds_byte_off;                                  // lds_addr
  g0[2] = (unsigned)ga;                                  // global_addr[31:0]
  g0[3] = ((unsigned)(ga >> 32) & 0x01ffffffu) | (2u << 30);  // addr[56:32] | type=2
  i32x8 g1;
  g1[0] = (int)((1u << 16) | (1u << 20) | (3u << 22) | (3u << 25)); // dsz=2B, pad 16dw->+4dw
  g1[1] = (int)((stride_elems & 0xffffu) << 16);                    // tensor_dim0 lo16
  g1[2] = (int)(((stride_elems >> 16) & 0xffffu) | ((tensor_rows & 0xffffu) << 16));
  g1[3] = (int)(((tensor_rows >> 16) & 0xffffu) | (32u << 16));     // tile_dim0 = 32
  g1[4] = tile_rows;                                                // tile_dim1 (tile_dim2=0)
  g1[5] = (int)stride_elems;                                        // tensor_dim0_stride lo32
  g1[6] = 0;
  g1[7] = 0;
  i32x4 z4; z4[0] = 0; z4[1] = 0; z4[2] = 0; z4[3] = 0;
#if defined(TDM_ARITY6)
  i32x8 z8; z8[0]=0; z8[1]=0; z8[2]=0; z8[3]=0; z8[4]=0; z8[5]=0; z8[6]=0; z8[7]=0;
  __builtin_amdgcn_tensor_load_to_lds(g0, g1, z4, z4, z8, 0);
#else
  __builtin_amdgcn_tensor_load_to_lds(g0, g1, z4, z4, 0);
#endif
}

// ---------------- small pipeline kernels ----------------

// fp32 -> bf16 with 32x32 tiled TRANSPOSE (per blockIdx.z matrix slice)
__global__ void cvt_transpose_kernel(const float* __restrict__ src, u16* __restrict__ dst,
                                     int R, int Cc) {
  __shared__ u16 tile[32][33];
  const int c0 = blockIdx.x * 32, r0 = blockIdx.y * 32;
  const size_t base = (size_t)blockIdx.z * R * Cc;
  const float* s = src + base;
  u16* d = dst + base;
  const int tx = threadIdx.x & 31, ty = threadIdx.x >> 5;  // 0..7
#pragma unroll
  for (int i = 0; i < 4; ++i) {
    const int r = ty + i * 8;
    tile[r][tx] = f2bf(s[(size_t)(r0 + r) * Cc + c0 + tx]);
  }
  __syncthreads();
#pragma unroll
  for (int i = 0; i < 4; ++i) {
    const int r = ty + i * 8;
    d[(size_t)(c0 + r) * R + r0 + tx] = tile[tx][r];
  }
}

__global__ void zero_kernel(unsigned int* __restrict__ p, size_t n) {
  size_t i = (size_t)blockIdx.x * blockDim.x + threadIdx.x;
  const size_t stride = (size_t)gridDim.x * blockDim.x;
  for (; i < n; i += stride) p[i] = 0u;
}

// one wave per token: logits, softmax over 8 experts, stable top-2
__global__ void router_kernel(const float* __restrict__ x, const float* __restrict__ rw,
                              const float* __restrict__ rb, float* __restrict__ probs,
                              int* __restrict__ eidx) {
  const int token = blockIdx.x * 8 + (threadIdx.x >> 5);
  const int lane  = threadIdx.x & 31;
  const float* xp = x + (size_t)token * NEMB;
  float acc[E_];
#pragma unroll
  for (int e = 0; e < E_; ++e) acc[e] = 0.f;
  for (int c = lane; c < NEMB; c += 32) {
    const float xv = xp[c];
    const float* rwc = rw + c * E_;
#pragma unroll
    for (int e = 0; e < E_; ++e) acc[e] += xv * rwc[e];
  }
#pragma unroll
  for (int off = 16; off > 0; off >>= 1)
#pragma unroll
    for (int e = 0; e < E_; ++e) acc[e] += __shfl_xor(acc[e], off, 32);
  if (lane == 0) {
    float p[E_]; float mx = -1e30f;
#pragma unroll
    for (int e = 0; e < E_; ++e) { p[e] = acc[e] + rb[e]; mx = fmaxf(mx, p[e]); }
    float sum = 0.f;
#pragma unroll
    for (int e = 0; e < E_; ++e) { p[e] = __expf(p[e] - mx); sum += p[e]; }
    const float inv = 1.f / sum;
#pragma unroll
    for (int e = 0; e < E_; ++e) p[e] *= inv;
    int i0 = 0;
#pragma unroll
    for (int e = 1; e < E_; ++e) if (p[e] > p[i0]) i0 = e;
    int i1 = (i0 == 0) ? 1 : 0;
#pragma unroll
    for (int e = 0; e < E_; ++e) if (e != i1 && e != i0 && p[e] > p[i1]) i1 = e;
    eidx[token * 2 + 0] = i0;  eidx[token * 2 + 1] = i1;
    probs[token * 2 + 0] = p[i0]; probs[token * 2 + 1] = p[i1];
  }
}

// one block per batch row; wave w == expert w scans (k-major, t-minor) flat order
__global__ void assign_kernel(const int* __restrict__ eidx, int* __restrict__ pos) {
  const int b = blockIdx.x;
  const int e = threadIdx.x >> 5;
  const int lane = threadIdx.x & 31;
  const int* eb = eidx + (size_t)b * NT_ * 2;
  int* pb = pos + (size_t)b * NT_ * 2;
  int count = 0;
  for (int k = 0; k < TOPK_; ++k) {
    for (int t0 = 0; t0 < NT_; t0 += 32) {
      const int t = t0 + lane;
      const bool mem = (eb[t * 2 + k] == e);
      const unsigned int m32 = (unsigned int)__ballot(mem);
      const int myofs = __popc(m32 & ((1u << lane) - 1u));
      if (mem) pb[t * 2 + k] = count + myofs;
      count += __popc(m32);
    }
  }
}

__global__ void scatter_kernel(const float* __restrict__ x, const int* __restrict__ eidx,
                               const int* __restrict__ pos, u16* __restrict__ Xb) {
  const int g = blockIdx.x;
  const int k = g & 1, token = g >> 1;
  const int b = token / NT_;
  const int e = eidx[token * 2 + k];
  const int p = pos[token * 2 + k];
  if (p >= CAP_) return;  // capacity drop
  const float* src = x + (size_t)token * NEMB;
  u16* dst = Xb + ((size_t)e * ME_ + (size_t)b * CAP_ + p) * NEMB;
  for (int c = threadIdx.x; c < NEMB; c += blockDim.x) dst[c] = f2bf(src[c]);
}

__global__ void gather_kernel(const float* __restrict__ EO, const int* __restrict__ eidx,
                              const int* __restrict__ pos, const float* __restrict__ probs,
                              float* __restrict__ y) {
  const int token = blockIdx.x;
  const int b = token / NT_;
  const int e0 = eidx[token * 2 + 0], e1 = eidx[token * 2 + 1];
  const int p0 = min(pos[token * 2 + 0], CAP_ - 1);
  const int p1 = min(pos[token * 2 + 1], CAP_ - 1);
  const float w0 = probs[token * 2 + 0], w1 = probs[token * 2 + 1];
  const float* r0 = EO + ((size_t)e0 * ME_ + (size_t)b * CAP_ + p0) * NEMB;
  const float* r1 = EO + ((size_t)e1 * ME_ + (size_t)b * CAP_ + p1) * NEMB;
  float* yo = y + (size_t)token * NEMB;
  for (int c = threadIdx.x; c < NEMB; c += blockDim.x) yo[c] = w0 * r0[c] + w1 * r1[c];
}

// ---------------- fused expert FFN: TDM staging + bf16 WMMA ----------------
// Block = 64-row M tile; 8 waves 2(M)x4(N), each wave 32x32 per output.
// Phase 1: h(64x2048)=silu(X@WgT^T+bg)*(X@WfT^T+bf) resident in LDS (bf16).
// Phase 2: EO tile = h @ WpT^T + bp, A-fragments straight from LDS h.
// All tile staging is TENSOR_LOAD_TO_LDS (wave 0 issues; double-buffered;
// s_wait_tensorcnt exploits in-order TDM completion).
// NOTE: no pointer arrays with LDS-derived initializers (lld cannot relocate
// addrspacecast constants in static initializers) — all buffer addresses are
// computed scalar-arithmetically from the smem base.

#define HSTR 2056                                   // 2048 + 8 pad ushorts
#define HS_BYTES (64 * HSTR * 2)                    // 263168
#define AS_BYTES (64 * 40 * 2)                      // 5120  (A stage, padded by TDM)
#define BS_BYTES (128 * 40 * 2)                     // 10240 (B^T stage, padded by TDM)
#define A_OFF  (HS_BYTES)
#define BG_OFF (HS_BYTES + 2 * AS_BYTES)
#define BF_OFF (HS_BYTES + 2 * AS_BYTES + 2 * BS_BYTES)
#define SMEM_BYTES (HS_BYTES + 2 * AS_BYTES + 4 * BS_BYTES)  // 314368 < 320KB/WGP

__global__ __launch_bounds__(256) void moe_expert_kernel(
    const u16* __restrict__ Xb, const u16* __restrict__ WgT, const u16* __restrict__ WfT,
    const u16* __restrict__ WpT, const float* __restrict__ bg, const float* __restrict__ bf_,
    const float* __restrict__ bp, float* __restrict__ EO) {
  extern __shared__ char smem[];
  u16* hS = (u16*)smem;
  const unsigned ldsBase = (unsigned)(uintptr_t)(void*)smem;  // LDS byte offset of smem

  const int tid  = threadIdx.x;
  const int lane = tid & 31;
  const int wave = tid >> 5;
  const int wm = wave >> 2;   // 0..1
  const int wn = wave & 3;    // 0..3
  const int e    = blockIdx.x / (ME_ / 64);
  const int row0 = (blockIdx.x % (ME_ / 64)) * 64;

  const u16* Xe   = Xb  + ((size_t)e * ME_ + row0) * NEMB;   // [64][1024] tile base
  const u16* WgTe = WgT + (size_t)e * NEMB * NHID;           // [NHID][NEMB]
  const u16* WfTe = WfT + (size_t)e * NEMB * NHID;           // [NHID][NEMB]
  const u16* WpTe = WpT + (size_t)e * NHID * NEMB;           // [NEMB][NHID]

  const f32x8 zv = {0.f, 0.f, 0.f, 0.f, 0.f, 0.f, 0.f, 0.f};

  // ---------------- phase 1 ----------------
  for (int nb = 0; nb < NHID / 128; ++nb) {
    const int n0 = nb * 128;
    f32x8 gacc[2][2], facc[2][2];
#pragma unroll
    for (int mi = 0; mi < 2; ++mi)
#pragma unroll
      for (int ni = 0; ni < 2; ++ni) { gacc[mi][ni] = zv; facc[mi][ni] = zv; }

    if (wave == 0) {  // prologue stage k0=0 -> buffers 0
      tdm_load_2d(Xe,                       ldsBase + A_OFF,  64, NEMB, 64);
      tdm_load_2d(WgTe + (size_t)n0 * NEMB, ldsBase + BG_OFF, 128, NEMB, 128);
      tdm_load_2d(WfTe + (size_t)n0 * NEMB, ldsBase + BF_OFF, 128, NEMB, 128);
    }

    for (int k0 = 0; k0 < NEMB; k0 += 32) {
      const int cur = (k0 >> 5) & 1;
      if (wave == 0) {
        if (k0 + 32 < NEMB) {
          const int nxt = cur ^ 1;
          tdm_load_2d(Xe + (k0 + 32),
                      ldsBase + A_OFF + nxt * AS_BYTES, 64, NEMB, 64);
          tdm_load_2d(WgTe + (size_t)n0 * NEMB + (k0 + 32),
                      ldsBase + BG_OFF + nxt * BS_BYTES, 128, NEMB, 128);
          tdm_load_2d(WfTe + (size_t)n0 * NEMB + (k0 + 32),
                      ldsBase + BF_OFF + nxt * BS_BYTES, 128, NEMB, 128);
          __builtin_amdgcn_s_wait_tensorcnt(3);  // in-order: current 3 tiles landed
        } else {
          __builtin_amdgcn_s_wait_tensorcnt(0);
        }
      }
      __syncthreads();

      const u16* aC  = (const u16*)(smem + A_OFF  + cur * AS_BYTES);
      const u16* bgC = (const u16*)(smem + BG_OFF + cur * BS_BYTES);
      const u16* bfC = (const u16*)(smem + BF_OFF + cur * BS_BYTES);

      bf16x16 afr[2], bgr[2], bfr[2];
#pragma unroll
      for (int mi = 0; mi < 2; ++mi) afr[mi] = load_a_frag(aC, wm * 32 + mi * 16, 0, 40, lane);
#pragma unroll
      for (int ni = 0; ni < 2; ++ni) {
        bgr[ni] = load_b_frag(bgC, wn * 32 + ni * 16, 40, lane);
        bfr[ni] = load_b_frag(bfC, wn * 32 + ni * 16, 40, lane);
      }
#pragma unroll
      for (int mi = 0; mi < 2; ++mi)
#pragma unroll
        for (int ni = 0; ni < 2; ++ni) {
          gacc[mi][ni] = wmma_bf16(afr[mi], bgr[ni], gacc[mi][ni]);
          facc[mi][ni] = wmma_bf16(afr[mi], bfr[ni], facc[mi][ni]);
        }
      __syncthreads();
    }

    // SwiGLU epilogue into LDS h (C/D layout: m = r + (lane/16)*8, n = lane%16)
#pragma unroll
    for (int mi = 0; mi < 2; ++mi)
#pragma unroll
      for (int ni = 0; ni < 2; ++ni) {
        const int nc = n0 + wn * 32 + ni * 16 + (lane & 15);
        const float bgv = bg[e * NHID + nc];
        const float bfv = bf_[e * NHID + nc];
#pragma unroll
        for (int r = 0; r < 8; ++r) {
          const int mr = wm * 32 + mi * 16 + ((lane >> 4) << 3) + r;
          const float gv = gacc[mi][ni][r] + bgv;
          const float fv = facc[mi][ni][r] + bfv;
          const float hv = (gv / (1.f + __expf(-gv))) * fv;
          hS[(size_t)mr * HSTR + nc] = f2bf(hv);
        }
      }
  }
  __syncthreads();

  // ---------------- phase 2 ----------------
  for (int nb = 0; nb < NEMB / 128; ++nb) {
    const int n0 = nb * 128;
    f32x8 oacc[2][2];
#pragma unroll
    for (int mi = 0; mi < 2; ++mi)
#pragma unroll
      for (int ni = 0; ni < 2; ++ni) oacc[mi][ni] = zv;

    if (wave == 0)
      tdm_load_2d(WpTe + (size_t)n0 * NHID, ldsBase + BG_OFF, 128, NHID, 128);

    for (int k0 = 0; k0 < NHID; k0 += 32) {
      const int cur = (k0 >> 5) & 1;
      if (wave == 0) {
        if (k0 + 32 < NHID) {
          tdm_load_2d(WpTe + (size_t)n0 * NHID + (k0 + 32),
                      ldsBase + BG_OFF + (cur ^ 1) * BS_BYTES, 128, NHID, 128);
          __builtin_amdgcn_s_wait_tensorcnt(1);
        } else {
          __builtin_amdgcn_s_wait_tensorcnt(0);
        }
      }
      __syncthreads();

      const u16* bpC = (const u16*)(smem + BG_OFF + cur * BS_BYTES);
      bf16x16 afr[2], bpr[2];
#pragma unroll
      for (int mi = 0; mi < 2; ++mi) afr[mi] = load_a_frag(hS, wm * 32 + mi * 16, k0, HSTR, lane);
#pragma unroll
      for (int ni = 0; ni < 2; ++ni) bpr[ni] = load_b_frag(bpC, wn * 32 + ni * 16, 40, lane);
#pragma unroll
      for (int mi = 0; mi < 2; ++mi)
#pragma unroll
        for (int ni = 0; ni < 2; ++ni) oacc[mi][ni] = wmma_bf16(afr[mi], bpr[ni], oacc[mi][ni]);
      __syncthreads();
    }

#pragma unroll
    for (int mi = 0; mi < 2; ++mi)
#pragma unroll
      for (int ni = 0; ni < 2; ++ni) {
        const int nc = n0 + wn * 32 + ni * 16 + (lane & 15);
        const float bpv = bp[e * NEMB + nc];
#pragma unroll
        for (int r = 0; r < 8; ++r) {
          const int mr = wm * 32 + mi * 16 + ((lane >> 4) << 3) + r;
          EO[((size_t)e * ME_ + row0 + mr) * NEMB + nc] = oacc[mi][ni][r] + bpv;
        }
      }
  }
}

// ---------------- host launcher ----------------
extern "C" void kernel_launch(void* const* d_in, const int* in_sizes, int n_in,
                              void* d_out, int out_size, void* d_ws, size_t ws_size,
                              hipStream_t stream) {
  (void)in_sizes; (void)n_in; (void)out_size; (void)ws_size;
  const float* x        = (const float*)d_in[0];
  const float* w_fc     = (const float*)d_in[1];
  const float* b_fc     = (const float*)d_in[2];
  const float* w_gate   = (const float*)d_in[3];
  const float* b_gate   = (const float*)d_in[4];
  const float* w_proj   = (const float*)d_in[5];
  const float* b_proj   = (const float*)d_in[6];
  const float* router_w = (const float*)d_in[7];
  const float* router_b = (const float*)d_in[8];
  float* y = (float*)d_out;

  char* ws = (char*)d_ws;
  size_t off = 0;
  auto take = [&](size_t bytes) -> void* {
    void* p = (void*)(ws + off);
    off += (bytes + 255) & ~(size_t)255;
    return p;
  };
  float* probs = (float*)take((size_t)NTOK * 2 * sizeof(float));
  int*   eidx  = (int*)  take((size_t)NTOK * 2 * sizeof(int));
  int*   pos   = (int*)  take((size_t)NTOK * 2 * sizeof(int));
  u16*   Xb    = (u16*)  take((size_t)E_ * ME_ * NEMB * sizeof(u16));
  u16*   WgT   = (u16*)  take((size_t)E_ * NEMB * NHID * sizeof(u16));
  u16*   WfT   = (u16*)  take((size_t)E_ * NEMB * NHID * sizeof(u16));
  u16*   WpT   = (u16*)  take((size_t)E_ * NHID * NEMB * sizeof(u16));
  float* EO    = (float*)take((size_t)E_ * ME_ * NEMB * sizeof(float));

  // bf16 conversion WITH transpose: WgT/WfT = [E][NHID][NEMB], WpT = [E][NEMB][NHID]
  cvt_transpose_kernel<<<dim3(NHID / 32, NEMB / 32, E_), 256, 0, stream>>>(w_gate, WgT, NEMB, NHID);
  cvt_transpose_kernel<<<dim3(NHID / 32, NEMB / 32, E_), 256, 0, stream>>>(w_fc,   WfT, NEMB, NHID);
  cvt_transpose_kernel<<<dim3(NEMB / 32, NHID / 32, E_), 256, 0, stream>>>(w_proj, WpT, NHID, NEMB);
  zero_kernel<<<2048, 256, 0, stream>>>((unsigned int*)Xb, (size_t)E_ * ME_ * NEMB / 2);

  router_kernel<<<NTOK / 8, 256, 0, stream>>>(x, router_w, router_b, probs, eidx);
  assign_kernel<<<NB_, 256, 0, stream>>>(eidx, pos);
  scatter_kernel<<<NTOK * TOPK_, 128, 0, stream>>>(x, eidx, pos, Xb);

  (void)hipFuncSetAttribute(reinterpret_cast<const void*>(moe_expert_kernel),
                            hipFuncAttributeMaxDynamicSharedMemorySize, SMEM_BYTES);
  moe_expert_kernel<<<E_ * (ME_ / 64), 256, SMEM_BYTES, stream>>>(
      Xb, WgT, WfT, WpT, b_gate, b_fc, b_proj, EO);

  gather_kernel<<<NTOK, 256, 0, stream>>>(EO, eidx, pos, probs, y);
}
